// GraphTransformer_3831110828330
// MI455X (gfx1250) — matrix operand order repaired and checked by reading, not compiled, not run on hardware
//
#include <hip/hip_runtime.h>
#include <hip/hip_bf16.h>

#define NNODES  50000
#define NEDGES  200000
#define IN_DIMC 256
#define HIDC    512
#define OUT_DIMC 384
#define NLAYERS 3

typedef __attribute__((ext_vector_type(16))) __bf16 v16bf;
typedef __attribute__((ext_vector_type(8)))  __bf16 v8bf;
typedef __attribute__((ext_vector_type(8)))  float  v8f;

union AFrag { v16bf v; v8bf h[2]; };

// ---------------------------------------------------------------------------
// Tiled GEMM with bias: C[M,Nout] = A[M,K] @ W[K,Nout] + bias
// fp32 global operands, bf16 staged in LDS (double-buffered),
// v_wmma_f32_16x16x32_bf16 compute.
// Block tile 128(M) x 64(N), 8 waves (4x2), wave tile 32x32 = 2x2 WMMA tiles.
// ---------------------------------------------------------------------------
#define BM 128
#define BN 64
#define BK 32

__global__ __launch_bounds__(256) void gemm_bias_wmma(
    const float* __restrict__ A, const float* __restrict__ W,
    const float* __restrict__ bias, float* __restrict__ C,
    int M, int K, int Nout)
{
    __shared__ __align__(64) __bf16 As[2][BM][BK];   // 2 x 8 KB
    __shared__ __align__(64) __bf16 Bs[2][BN][BK];   // 2 x 4 KB

    const int tid  = threadIdx.x;
    const int lane = tid & 31;
    const int wave = tid >> 5;
    const int bm = blockIdx.x * BM;
    const int bn = blockIdx.y * BN;
    const int wm = (wave & 3) * 32;   // wave M offset in block
    const int wn = (wave >> 2) * 32;  // wave N offset in block

    const int lrow  = lane & 15;          // M (A) / N (B) index within tile
    const int khalf = (lane >> 4) * 16;   // A: K chunk offset per lane-half
    const int kb16  = (lane >> 4) * 16;   // B: K chunk offset per lane-half

    // ---- staging assignments (fixed for whole kernel) ----
    const int rA    = tid >> 1;           // A: 2 threads per row
    const int cbase = (tid & 1) * 16;
    {
    }
    const int grow  = bm + rA;
    const int arow  = grow < M ? grow : (M - 1);  // clamp keeps EXEC uniform
    const float* __restrict__ Arow = A + (size_t)arow * K;
    const int nW  = tid & 63;             // B: column within tile
    const int kbW = (tid >> 6) * 8;       // B: 8 k-values per thread

    v8f acc[2][2] = {};
    float4 afr[4];
    float  bfr[8];

    const int ntiles = K / BK;

    // ---- prologue: fetch tile 0 and stage into buffer 0 ----
#pragma unroll
    for (int i = 0; i < 4; ++i)
        afr[i] = *(const float4*)(Arow + cbase + i * 4);
#pragma unroll
    for (int i = 0; i < 8; ++i)
        bfr[i] = W[(size_t)(kbW + i) * Nout + bn + nW];
#pragma unroll
    for (int i = 0; i < 4; ++i) {
        int c = cbase + i * 4;
        As[0][rA][c + 0] = (__bf16)afr[i].x;  As[0][rA][c + 1] = (__bf16)afr[i].y;
        As[0][rA][c + 2] = (__bf16)afr[i].z;  As[0][rA][c + 3] = (__bf16)afr[i].w;
    }
#pragma unroll
    for (int i = 0; i < 8; ++i)
        Bs[0][nW][kbW + i] = (__bf16)bfr[i];
    __syncthreads();

    for (int t = 0; t < ntiles; ++t) {
        const int  cur = t & 1, nxt = cur ^ 1;
        const bool have_next = (t + 1) < ntiles;     // uniform branch

        // ---- issue global loads for next tile (overlaps WMMA below) ----
        if (have_next) {
            const int k0 = (t + 1) * BK;
#pragma unroll
            for (int i = 0; i < 4; ++i)
                afr[i] = *(const float4*)(Arow + k0 + cbase + i * 4);
#pragma unroll
            for (int i = 0; i < 8; ++i)
                bfr[i] = W[(size_t)(k0 + kbW + i) * Nout + bn + nW];
        }

        // ---- compute from current buffer ----
#pragma unroll
        for (int ti = 0; ti < 2; ++ti) {
            AFrag a;
            int ar = wm + ti * 16 + lrow;
            a.h[0] = *(const v8bf*)&As[cur][ar][khalf];       // K = khalf..+7
            a.h[1] = *(const v8bf*)&As[cur][ar][khalf + 8];   // K = khalf+8..+15
#pragma unroll
            for (int tj = 0; tj < 2; ++tj) {
                AFrag b;
                int bc = wn + tj * 16 + lrow;
                b.v = *(const v16bf*)&Bs[cur][bc][kb16];       // 16 contiguous K
                acc[ti][tj] = __builtin_amdgcn_wmma_f32_16x16x32_bf16(
                    false, a.v, false, b.v, (short)0, acc[ti][tj], false, false);
            }
        }

        // ---- stage next tile into the other buffer ----
        if (have_next) {
#pragma unroll
            for (int i = 0; i < 4; ++i) {
                int c = cbase + i * 4;
                As[nxt][rA][c + 0] = (__bf16)afr[i].x;
                As[nxt][rA][c + 1] = (__bf16)afr[i].y;
                As[nxt][rA][c + 2] = (__bf16)afr[i].z;
                As[nxt][rA][c + 3] = (__bf16)afr[i].w;
            }
#pragma unroll
            for (int i = 0; i < 8; ++i)
                Bs[nxt][nW][kbW + i] = (__bf16)bfr[i];
        }
        __syncthreads();
    }

    // ---- store D + bias; C-layout: VGPR r -> row r (+8 for upper lane-half) ----
    const int rowadd = (lane >> 4) * 8;
    if (bm + BM <= M) {
        // full tile: straight-line stores, no exec-mask churn
#pragma unroll
        for (int ti = 0; ti < 2; ++ti) {
#pragma unroll
            for (int tj = 0; tj < 2; ++tj) {
                int col = bn + wn + tj * 16 + lrow;
                float bv = bias[col];
                float* cp = C + (size_t)(bm + wm + ti * 16 + rowadd) * Nout + col;
#pragma unroll
                for (int r = 0; r < 8; ++r)
                    cp[(size_t)r * Nout] = acc[ti][tj][r] + bv;
            }
        }
    } else {
        // last partial M tile only
#pragma unroll
        for (int ti = 0; ti < 2; ++ti) {
#pragma unroll
            for (int tj = 0; tj < 2; ++tj) {
                int col = bn + wn + tj * 16 + lrow;
                float bv = bias[col];
#pragma unroll
                for (int r = 0; r < 8; ++r) {
                    int row = bm + wm + ti * 16 + r + rowadd;
                    if (row < M)
                        C[(size_t)row * Nout + col] = acc[ti][tj][r] + bv;
                }
            }
        }
    }
}

// ---------------------------------------------------------------------------
// Ordered-uint encoding for float atomicMax
// ---------------------------------------------------------------------------
__device__ __forceinline__ unsigned enc_f32(float f) {
    unsigned b = __float_as_uint(f);
    return (b & 0x80000000u) ? ~b : (b | 0x80000000u);
}
__device__ __forceinline__ float dec_f32(unsigned e) {
    unsigned b = (e & 0x80000000u) ? (e & 0x7fffffffu) : ~e;
    return __uint_as_float(b);
}

// init segment buffers: mEnc=0 (== -inf encoded floor), s=0, agg=0
__global__ void init_segments(unsigned* __restrict__ mEnc, float* __restrict__ ssum,
                              float* __restrict__ agg, int nNH, int nAgg)
{
    int i = blockIdx.x * blockDim.x + threadIdx.x;
    if (i < nNH) { mEnc[i] = 0u; ssum[i] = 0.f; }
    if (i < nAgg) agg[i] = 0.f;
}

// ---------------------------------------------------------------------------
// One wave per edge: per-head q.k dot, store logits, segment-max via atomicMax
// ---------------------------------------------------------------------------
__global__ __launch_bounds__(256) void edge_logits(
    const int* __restrict__ ei, const float* __restrict__ q,
    const float* __restrict__ k, float* __restrict__ logits,
    unsigned* __restrict__ mEnc)
{
    int e    = (blockIdx.x * blockDim.x + threadIdx.x) >> 5;
    int lane = threadIdx.x & 31;
    if (e >= NEDGES) return;
    int s = ei[e];
    int d = ei[NEDGES + e];
    const float4* qp = (const float4*)(q + (size_t)d * HIDC);
    const float4* kp = (const float4*)(k + (size_t)s * HIDC);
    float acc = 0.f;
#pragma unroll
    for (int i = 0; i < 4; ++i) {                 // lane covers 16 floats
        float4 a = qp[lane * 4 + i];
        float4 b = kp[lane * 4 + i];
        acc += a.x * b.x + a.y * b.y + a.z * b.z + a.w * b.w;
    }
    // reduce within 8-lane head groups (128 dims / head)
    acc += __shfl_xor(acc, 1, 32);
    acc += __shfl_xor(acc, 2, 32);
    acc += __shfl_xor(acc, 4, 32);
    if ((lane & 7) == 0) {
        int h = lane >> 3;
        float lg = acc * 0.08838834764831845f;    // 1/sqrt(128)
        logits[(size_t)e * 4 + h] = lg;
        atomicMax(&mEnc[(size_t)d * 4 + h], enc_f32(lg));
    }
}

// exp(logit - m[dst]) in place; segment-sum via atomicAdd
__global__ void edge_exp(const int* __restrict__ ei, float* __restrict__ logits,
                         const unsigned* __restrict__ mEnc, float* __restrict__ ssum)
{
    int idx = blockIdx.x * blockDim.x + threadIdx.x;
    if (idx >= NEDGES * 4) return;
    int e = idx >> 2, h = idx & 3;
    int d = ei[NEDGES + e];
    float m  = dec_f32(mEnc[(size_t)d * 4 + h]);
    float ev = __expf(logits[idx] - m);
    logits[idx] = ev;
    atomicAdd(&ssum[(size_t)d * 4 + h], ev);
}

// one 128-thread block per edge: agg[dst] += alpha * v[src]
__global__ __launch_bounds__(128) void edge_scatter(
    const int* __restrict__ ei, const float* __restrict__ alpha,
    const float* __restrict__ ssum, const float* __restrict__ v,
    float* __restrict__ agg)
{
    int e = blockIdx.x;
    int t = threadIdx.x;                 // covers 4 floats each -> 512
    int s = ei[e], d = ei[NEDGES + e];
    int h = t >> 5;                      // element (t*4)/128
    float a = alpha[(size_t)e * 4 + h] /
              fmaxf(ssum[(size_t)d * 4 + h], 1e-16f);
    float4 vv = ((const float4*)(v + (size_t)s * HIDC))[t];
    float* ag = agg + (size_t)d * HIDC + t * 4;
    atomicAdd(&ag[0], a * vv.x);
    atomicAdd(&ag[1], a * vv.y);
    atomicAdd(&ag[2], a * vv.z);
    atomicAdd(&ag[3], a * vv.w);
}

// ---------------------------------------------------------------------------
// h = LayerNorm(h + agg + hs) * g + b   (one wave per 512-wide row, in place)
// ---------------------------------------------------------------------------
__global__ __launch_bounds__(256) void add_layernorm(
    float* __restrict__ h, const float* __restrict__ agg,
    const float* __restrict__ hs, const float* __restrict__ g,
    const float* __restrict__ b)
{
    int row  = (blockIdx.x * blockDim.x + threadIdx.x) >> 5;
    int lane = threadIdx.x & 31;
    if (row >= NNODES) return;
    float*       hp = h   + (size_t)row * HIDC;
    const float* ap = agg + (size_t)row * HIDC;
    const float* sp = hs  + (size_t)row * HIDC;
    float vals[16];
    float sum = 0.f;
#pragma unroll
    for (int i = 0; i < 16; ++i) {
        int c = i * 32 + lane;
        float t = hp[c] + ap[c] + sp[c];
        vals[i] = t;  sum += t;
    }
#pragma unroll
    for (int off = 16; off >= 1; off >>= 1) sum += __shfl_xor(sum, off, 32);
    float mu = sum * (1.0f / HIDC);
    float var = 0.f;
#pragma unroll
    for (int i = 0; i < 16; ++i) { float dx = vals[i] - mu; var += dx * dx; }
#pragma unroll
    for (int off = 16; off >= 1; off >>= 1) var += __shfl_xor(var, off, 32);
    var *= (1.0f / HIDC);
    float inv = rsqrtf(var + 1e-5f);
#pragma unroll
    for (int i = 0; i < 16; ++i) {
        int c = i * 32 + lane;
        hp[c] = (vals[i] - mu) * inv * g[c] + b[c];
    }
}

// ---------------------------------------------------------------------------
extern "C" void kernel_launch(void* const* d_in, const int* in_sizes, int n_in,
                              void* d_out, int out_size, void* d_ws, size_t ws_size,
                              hipStream_t stream)
{
    (void)in_sizes; (void)n_in; (void)out_size; (void)ws_size;

    const float* x    = (const float*)d_in[0];
    const int*   ei   = (const int*)  d_in[1];
    const float* W_in = (const float*)d_in[2];
    const float* b_in = (const float*)d_in[3];
    const float* Wq   = (const float*)d_in[4];
    const float* bq   = (const float*)d_in[5];
    const float* Wk   = (const float*)d_in[6];
    const float* bk   = (const float*)d_in[7];
    const float* Wv   = (const float*)d_in[8];
    const float* bv   = (const float*)d_in[9];
    const float* Ws   = (const float*)d_in[10];
    const float* bs   = (const float*)d_in[11];
    const float* ln_g = (const float*)d_in[12];
    const float* ln_b = (const float*)d_in[13];
    const float* W_out= (const float*)d_in[14];
    const float* b_out= (const float*)d_in[15];

    const size_t NH = (size_t)NNODES * HIDC;
    float* ws   = (float*)d_ws;
    float* h    = ws;
    float* q    = ws + 1 * NH;
    float* k    = ws + 2 * NH;
    float* v    = ws + 3 * NH;
    float* hs   = ws + 4 * NH;
    float* agg  = ws + 5 * NH;
    float*    logits = ws + 6 * NH;                               // E*4
    unsigned* mEnc   = (unsigned*)(logits + (size_t)NEDGES * 4);  // N*4
    float*    ssum   = (float*)(mEnc + (size_t)NNODES * 4);       // N*4

    const dim3 blk(256);
    const int  mTiles = (NNODES + BM - 1) / BM;           // 391

    // h = x @ W_in + b_in
    gemm_bias_wmma<<<dim3(mTiles, HIDC / BN), blk, 0, stream>>>(
        x, W_in, b_in, h, NNODES, IN_DIMC, HIDC);

    for (int l = 0; l < NLAYERS; ++l) {
        const size_t wOff = (size_t)l * HIDC * HIDC;
        const size_t bOff = (size_t)l * HIDC;
        gemm_bias_wmma<<<dim3(mTiles, HIDC / BN), blk, 0, stream>>>(
            h, Wq + wOff, bq + bOff, q, NNODES, HIDC, HIDC);
        gemm_bias_wmma<<<dim3(mTiles, HIDC / BN), blk, 0, stream>>>(
            h, Wk + wOff, bk + bOff, k, NNODES, HIDC, HIDC);
        gemm_bias_wmma<<<dim3(mTiles, HIDC / BN), blk, 0, stream>>>(
            h, Wv + wOff, bv + bOff, v, NNODES, HIDC, HIDC);
        gemm_bias_wmma<<<dim3(mTiles, HIDC / BN), blk, 0, stream>>>(
            h, Ws + wOff, bs + bOff, hs, NNODES, HIDC, HIDC);

        init_segments<<<(int)((NH + 255) / 256), blk, 0, stream>>>(
            mEnc, ssum, agg, NNODES * 4, (int)NH);

        edge_logits<<<(NEDGES * 32 + 255) / 256, blk, 0, stream>>>(
            ei, q, k, logits, mEnc);
        edge_exp<<<(NEDGES * 4 + 255) / 256, blk, 0, stream>>>(
            ei, logits, mEnc, ssum);
        edge_scatter<<<NEDGES, dim3(128), 0, stream>>>(
            ei, logits, ssum, v, agg);

        add_layernorm<<<(NNODES * 32 + 255) / 256, blk, 0, stream>>>(
            h, agg, hs, ln_g + bOff, ln_b + bOff);
    }

    // out = h @ W_out + b_out
    gemm_bias_wmma<<<dim3(mTiles, OUT_DIMC / BN), blk, 0, stream>>>(
        h, W_out, b_out, (float*)d_out, NNODES, HIDC, OUT_DIMC);
}
